// MultiHeadAttention_88965952569751
// MI455X (gfx1250) — matrix operand order repaired and checked by reading, not compiled, not run on hardware
//
#include <hip/hip_runtime.h>
#include <hip/hip_bf16.h>
#include <stdint.h>

// ---------------- problem constants ----------------
#define D_MODELc 1024
#define HEADSc   16
#define DKc      64
#define SEQc     2048
#define BATCHc   2
#define MTOT     (BATCHc * SEQc)   // 4096 rows total

typedef __attribute__((ext_vector_type(16))) __bf16 v16bf;
typedef __attribute__((ext_vector_type(8)))  float  v8f;
typedef __attribute__((ext_vector_type(4)))  int    i32x4;
typedef unsigned short u16;
typedef unsigned int   u32;

// ---------------- CDNA5 async-to-LDS (ASYNCcnt) ----------------
#if defined(__has_builtin)
#if __has_builtin(__builtin_amdgcn_global_load_async_to_lds_b128)
#define HAVE_ASYNC_BUILTIN 1
#endif
#if __has_builtin(__builtin_amdgcn_s_wait_asynccnt)
#define HAVE_ASYNCCNT_BUILTIN 1
#endif
#endif

__device__ __forceinline__ void async_copy16(const u16* gsrc, u16* ldst) {
#ifdef HAVE_ASYNC_BUILTIN
    __builtin_amdgcn_global_load_async_to_lds_b128(
        (__attribute__((address_space(1))) i32x4*)gsrc,
        (__attribute__((address_space(3))) i32x4*)ldst, 0, 0);
#else
    u32 laddr = (u32)(size_t)(__attribute__((address_space(3))) u16*)ldst;
    asm volatile("global_load_async_to_lds_b128 %0, %1, off"
                 :: "v"(laddr), "v"(gsrc) : "memory");
#endif
}
__device__ __forceinline__ void wait_async0() {
#ifdef HAVE_ASYNCCNT_BUILTIN
    __builtin_amdgcn_s_wait_asynccnt(0);
#else
    asm volatile("s_wait_asynccnt 0" ::: "memory");
#endif
}

// ---------------- bf16 helpers ----------------
struct B32x8 { uint4 lo, hi; };

__device__ __forceinline__ v16bf frag16(uint4 lo, uint4 hi) {
    B32x8 t; t.lo = lo; t.hi = hi;
    return __builtin_bit_cast(v16bf, t);
}
__device__ __forceinline__ u16 f2bf(float f) {
    u32 u = __float_as_uint(f);
    u32 r = u + 0x7FFFu + ((u >> 16) & 1u);   // round-to-nearest-even
    return (u16)(r >> 16);
}
__device__ __forceinline__ u32 pk2(float a, float b) {
    return (u32)f2bf(a) | ((u32)f2bf(b) << 16);
}
__device__ __forceinline__ uint4 pk8(float4 a, float4 b) {
    uint4 r;
    r.x = pk2(a.x, a.y); r.y = pk2(a.z, a.w);
    r.z = pk2(b.x, b.y); r.w = pk2(b.z, b.w);
    return r;
}

// =====================================================================
// One-shot f32 -> bf16 conversion (weights; each element converted once)
// =====================================================================
__global__ __launch_bounds__(256)
void cvt_bf16_kernel(const float* __restrict__ src, u16* __restrict__ dst, int n)
{
    const int i = (blockIdx.x * 256 + threadIdx.x) * 8;
    if (i + 8 <= n) {
        float4 a = *(const float4*)(src + i);
        float4 b = *(const float4*)(src + i + 4);
        *(uint4*)(dst + i) = pk8(a, b);
    }
}

// =====================================================================
// Projection GEMM: out[m][n] = sum_k A[m][k] * W[n][k] + bias[n]
// A: f32 row-major [MTOT][1024], Wb: bf16 [1024][1024] (pre-converted)
// VSTORE==0: out row-major bf16 [MTOT][1024]  (Q, K)
// VSTORE==1: out transposed per head Vt[b][h][d][s] bf16  (V)
// Block: 256 threads = 8 waves, tile 128(M) x 64(N), k-step 32.
// W tile (64x32 bf16 = 4KB) staged via GLOBAL_LOAD_ASYNC_TO_LDS_B128.
// =====================================================================
template<int VSTORE>
__global__ __launch_bounds__(256)
void proj_kernel(const float* __restrict__ A, const u16* __restrict__ Wb,
                 const float* __restrict__ bias, u16* __restrict__ out)
{
    __shared__ __align__(16) u16 sW[64 * 32];   // [n_local][k_local] bf16

    const int tid  = threadIdx.x;
    const int lane = tid & 31;
    const int wave = tid >> 5;
    const int n0   = blockIdx.x * 64;
    const int m0   = blockIdx.y * 128 + wave * 16;
    const int mrow = lane & 15;
    const int half = lane >> 4;

    // per-thread staging chunk: 256 chunks of 16B cover 64x32 bf16
    const int c_nl = tid >> 2;          // 0..63  (row of W tile)
    const int c_kq = (tid & 3) * 8;     // 0,8,16,24 (u16 offset)
    const u16* wbase = Wb + (size_t)(n0 + c_nl) * D_MODELc + c_kq;
    u16* lbase = &sW[c_nl * 32 + c_kq];

    v8f zero = {};
    v8f acc[4];
#pragma unroll
    for (int t = 0; t < 4; ++t) acc[t] = zero;

    for (int k0 = 0; k0 < D_MODELc; k0 += 32) {
        async_copy16(wbase + k0, lbase);   // 1 async b128 per thread
        wait_async0();
        __syncthreads();

        // A fragment (16x32, ISA A-layout): two contiguous 8-elem chunks
        const float* ap = A + (size_t)(m0 + mrow) * D_MODELc + k0 + half * 8;
        float4 a0 = *(const float4*)ap;
        float4 a1 = *(const float4*)(ap + 4);
        float4 a2 = *(const float4*)(ap + 16);
        float4 a3 = *(const float4*)(ap + 20);
        v16bf af = frag16(pk8(a0, a1), pk8(a2, a3));

#pragma unroll
        for (int t = 0; t < 4; ++t) {
            const u16* bp = &sW[(t * 16 + mrow) * 32 + half * 8];
            v16bf bf = frag16(*(const uint4*)bp, *(const uint4*)(bp + 16));
            acc[t] = __builtin_amdgcn_wmma_f32_16x16x32_bf16(
                false, af, false, bf, (short)0, acc[t], false, false);
        }
        __syncthreads();   // all reads done before next async stage
    }

#pragma unroll
    for (int t = 0; t < 4; ++t) {
        const int n = n0 + t * 16 + mrow;
        const float bvv = bias[n];
#pragma unroll
        for (int r = 0; r < 8; ++r) {
            const int m = m0 + r + half * 8;
            const float val = acc[t][r] + bvv;
            if (VSTORE == 0) {
                out[(size_t)m * D_MODELc + n] = f2bf(val);
            } else {
                const int hh = n >> 6, d = n & 63;
                const int bb = m >> 11, s = m & (SEQc - 1);
                out[(((size_t)bb * HEADSc + hh) * DKc + d) * SEQc + s] = f2bf(val);
            }
        }
    }
}

// =====================================================================
// Output projection: out[m][n] = sum_k Abf[m][k]*Wb[n][k] + bias[n], f32 out
// =====================================================================
__global__ __launch_bounds__(256)
void outproj_kernel(const u16* __restrict__ A, const u16* __restrict__ Wb,
                    const float* __restrict__ bias, float* __restrict__ out)
{
    __shared__ __align__(16) u16 sW[64 * 32];

    const int tid  = threadIdx.x;
    const int lane = tid & 31;
    const int wave = tid >> 5;
    const int n0   = blockIdx.x * 64;
    const int m0   = blockIdx.y * 128 + wave * 16;
    const int mrow = lane & 15;
    const int half = lane >> 4;

    const int c_nl = tid >> 2;
    const int c_kq = (tid & 3) * 8;
    const u16* wbase = Wb + (size_t)(n0 + c_nl) * D_MODELc + c_kq;
    u16* lbase = &sW[c_nl * 32 + c_kq];

    v8f zero = {};
    v8f acc[4];
#pragma unroll
    for (int t = 0; t < 4; ++t) acc[t] = zero;

    for (int k0 = 0; k0 < D_MODELc; k0 += 32) {
        async_copy16(wbase + k0, lbase);
        wait_async0();
        __syncthreads();

        const u16* ap = A + (size_t)(m0 + mrow) * D_MODELc + k0 + half * 8;
        v16bf af = frag16(*(const uint4*)ap, *(const uint4*)(ap + 16));

#pragma unroll
        for (int t = 0; t < 4; ++t) {
            const u16* bp = &sW[(t * 16 + mrow) * 32 + half * 8];
            v16bf bf = frag16(*(const uint4*)bp, *(const uint4*)(bp + 16));
            acc[t] = __builtin_amdgcn_wmma_f32_16x16x32_bf16(
                false, af, false, bf, (short)0, acc[t], false, false);
        }
        __syncthreads();
    }

#pragma unroll
    for (int t = 0; t < 4; ++t) {
        const int n = n0 + t * 16 + mrow;
        const float bvv = bias[n];
#pragma unroll
        for (int r = 0; r < 8; ++r) {
            const int m = m0 + r + half * 8;
            out[(size_t)m * D_MODELc + n] = acc[t][r] + bvv;
        }
    }
}

// =====================================================================
// Flash attention. Block = 4 waves sharing one (b,h), 4 consecutive
// 16-row q-blocks. K tile (32x64) and V tile (64x32) staged once per
// block per j-step via async-to-LDS and shared by all 4 waves (4x less
// L2 traffic than per-wave loads). Online softmax, f32 accumulators.
// Qp/Kp: bf16 [B*S][1024], Vt: bf16 [B][H][DK][S], Ao: bf16 [B*S][1024].
// =====================================================================
__global__ __launch_bounds__(128)
void attn_kernel(const u16* __restrict__ Qp, const u16* __restrict__ Kp,
                 const u16* __restrict__ Vt, u16* __restrict__ Ao)
{
    __shared__ __align__(16) u16 sK[32 * 64];      // [j_local][d]  4KB
    __shared__ __align__(16) u16 sV[64 * 32];      // [d][j_local]  4KB
    __shared__ __align__(16) u16 sP[4][16 * 32];   // per-wave P     4KB

    const int tid  = threadIdx.x;
    const int lane = tid & 31;
    const int wave = tid >> 5;
    const int bid  = blockIdx.x;                   // 0..1023
    const int qt   = bid & 31;                     // S/64 = 32 q-tiles
    const int h    = (bid >> 5) & 15;
    const int b    = bid >> 9;
    const int q0   = qt * 64 + wave * 16;
    const int mrow = lane & 15;
    const int half = lane >> 4;

    // ---- staging chunk maps (16B chunks, 2 per thread per tile) ----
    // K tile: 32 rows x 128B -> 256 chunks: row = c>>3, off = (c&7)*8
    const int kc0 = tid, kc1 = tid + 128;
    const u16* kg0 = Kp + (size_t)(b * SEQc + (kc0 >> 3)) * D_MODELc + h * DKc + (kc0 & 7) * 8;
    const u16* kg1 = Kp + (size_t)(b * SEQc + (kc1 >> 3)) * D_MODELc + h * DKc + (kc1 & 7) * 8;
    u16* kl0 = &sK[(kc0 >> 3) * 64 + (kc0 & 7) * 8];
    u16* kl1 = &sK[(kc1 >> 3) * 64 + (kc1 & 7) * 8];
    // V tile: 64 rows x 64B -> 256 chunks: d = c>>2, off = (c&3)*8
    const u16* vg0 = Vt + ((size_t)(b * HEADSc + h) * DKc + (kc0 >> 2)) * SEQc + (kc0 & 3) * 8;
    const u16* vg1 = Vt + ((size_t)(b * HEADSc + h) * DKc + (kc1 >> 2)) * SEQc + (kc1 & 3) * 8;
    u16* vl0 = &sV[(kc0 >> 2) * 32 + (kc0 & 3) * 8];
    u16* vl1 = &sV[(kc1 >> 2) * 32 + (kc1 & 3) * 8];

    // ---- Q fragments (16x64 -> two 16x32 A-fragments), loaded once ----
    const u16* qrow = Qp + (size_t)(b * SEQc + q0 + mrow) * D_MODELc + h * DKc + half * 8;
    v16bf qf0 = frag16(*(const uint4*)qrow,        *(const uint4*)(qrow + 16));
    v16bf qf1 = frag16(*(const uint4*)(qrow + 32), *(const uint4*)(qrow + 48));

    v8f zero = {};
    v8f oacc[4];
#pragma unroll
    for (int t = 0; t < 4; ++t) oacc[t] = zero;
    float mst[8], lst[8];
#pragma unroll
    for (int r = 0; r < 8; ++r) { mst[r] = -1e30f; lst[r] = 0.0f; }

    u16* pbuf = sP[wave];

    for (int j0 = 0; j0 < SEQc; j0 += 32) {
        // ---- async-stage K and V tiles for this j-slab ----
        const size_t kjump = (size_t)j0 * D_MODELc;   // K advances by rows
        async_copy16(kg0 + kjump, kl0);
        async_copy16(kg1 + kjump, kl1);
        async_copy16(vg0 + j0, vl0);                  // V advances along s
        async_copy16(vg1 + j0, vl1);
        wait_async0();
        __syncthreads();

        // ---- scores: S[16x32] = Q[16x64] @ K^T, two 16-col tiles ----
        v8f sc[2]; sc[0] = zero; sc[1] = zero;
#pragma unroll
        for (int jt = 0; jt < 2; ++jt) {
            const u16* kr = &sK[(jt * 16 + mrow) * 64 + half * 8];
            v16bf kf0 = frag16(*(const uint4*)kr,        *(const uint4*)(kr + 16));
            v16bf kf1 = frag16(*(const uint4*)(kr + 32), *(const uint4*)(kr + 48));
            sc[jt] = __builtin_amdgcn_wmma_f32_16x16x32_bf16(
                false, qf0, false, kf0, (short)0, sc[jt], false, false);
            sc[jt] = __builtin_amdgcn_wmma_f32_16x16x32_bf16(
                false, qf1, false, kf1, (short)0, sc[jt], false, false);
        }

        // ---- online softmax over this 32-wide j-slab ----
        float p0[8], p1[8];
#pragma unroll
        for (int r = 0; r < 8; ++r) {
            float s0 = sc[0][r] * 0.125f;     // 1/sqrt(Dk)
            float s1 = sc[1][r] * 0.125f;
            float mx = fmaxf(s0, s1);
#pragma unroll
            for (int off = 1; off < 16; off <<= 1)
                mx = fmaxf(mx, __shfl_xor(mx, off, 16));
            const float mnew = fmaxf(mst[r], mx);
            const float corr = __expf(mst[r] - mnew);
            const float e0 = __expf(s0 - mnew);
            const float e1 = __expf(s1 - mnew);
            float rs = e0 + e1;
#pragma unroll
            for (int off = 1; off < 16; off <<= 1)
                rs += __shfl_xor(rs, off, 16);
            lst[r] = lst[r] * corr + rs;
            mst[r] = mnew;
#pragma unroll
            for (int t = 0; t < 4; ++t) oacc[t][r] *= corr;
            p0[r] = e0; p1[r] = e1;
        }

        // ---- P: C-layout -> row-major bf16 in per-wave LDS buffer ----
#pragma unroll
        for (int r = 0; r < 8; ++r) {
            const int m = r + half * 8;
            pbuf[m * 32 + mrow]      = f2bf(p0[r]);
            pbuf[m * 32 + 16 + mrow] = f2bf(p1[r]);
        }
        asm volatile("s_wait_dscnt 0" ::: "memory");   // same-wave LDS RAW

        // ---- reload P as A-fragment, accumulate O += P @ V ----
        const u16* pp = pbuf + mrow * 32 + half * 8;
        v16bf pf = frag16(*(const uint4*)pp, *(const uint4*)(pp + 16));
#pragma unroll
        for (int t = 0; t < 4; ++t) {
            const u16* vp = &sV[(t * 16 + mrow) * 32 + half * 8];
            v16bf vf = frag16(*(const uint4*)vp, *(const uint4*)(vp + 16));
            oacc[t] = __builtin_amdgcn_wmma_f32_16x16x32_bf16(
                false, pf, false, vf, (short)0, oacc[t], false, false);
        }
        __syncthreads();   // all sK/sV reads done before next async stage
    }

    // ---- normalize and store bf16 ----
#pragma unroll
    for (int t = 0; t < 4; ++t) {
#pragma unroll
        for (int r = 0; r < 8; ++r) {
            const float inv = 1.0f / lst[r];
            const int m = b * SEQc + q0 + r + half * 8;
            Ao[(size_t)m * D_MODELc + h * DKc + t * 16 + mrow] = f2bf(oacc[t][r] * inv);
        }
    }
}

// =====================================================================
extern "C" void kernel_launch(void* const* d_in, const int* in_sizes, int n_in,
                              void* d_out, int out_size, void* d_ws, size_t ws_size,
                              hipStream_t stream)
{
    (void)in_sizes; (void)n_in; (void)out_size; (void)ws_size;
    const float* v  = (const float*)d_in[0];
    const float* k  = (const float*)d_in[1];
    const float* q  = (const float*)d_in[2];
    const float* Wv = (const float*)d_in[3];
    const float* bv = (const float*)d_in[4];
    const float* Wk = (const float*)d_in[5];
    const float* bk = (const float*)d_in[6];
    const float* Wq = (const float*)d_in[7];
    const float* bq = (const float*)d_in[8];
    const float* Wo = (const float*)d_in[9];
    const float* bo = (const float*)d_in[10];
    float* out = (float*)d_out;

    // workspace layout (u16 units):
    //   Qp, Kp, Vt, Ao : 4 x [4096*1024]  (8 MB each)
    //   Wqb, Wkb, Wvb, Wob : 4 x [1024*1024] (2 MB each)
    u16* Qp  = (u16*)d_ws;
    u16* Kp  = Qp  + (size_t)MTOT * D_MODELc;
    u16* Vt  = Kp  + (size_t)MTOT * D_MODELc;
    u16* Ao  = Vt  + (size_t)MTOT * D_MODELc;
    u16* Wqb = Ao  + (size_t)MTOT * D_MODELc;
    u16* Wkb = Wqb + (size_t)D_MODELc * D_MODELc;
    u16* Wvb = Wkb + (size_t)D_MODELc * D_MODELc;
    u16* Wob = Wvb + (size_t)D_MODELc * D_MODELc;

    const int WN = D_MODELc * D_MODELc;
    const int cvtBlocks = WN / (256 * 8);   // 512
    cvt_bf16_kernel<<<cvtBlocks, 256, 0, stream>>>(Wq, Wqb, WN);
    cvt_bf16_kernel<<<cvtBlocks, 256, 0, stream>>>(Wk, Wkb, WN);
    cvt_bf16_kernel<<<cvtBlocks, 256, 0, stream>>>(Wv, Wvb, WN);
    cvt_bf16_kernel<<<cvtBlocks, 256, 0, stream>>>(Wo, Wob, WN);

    dim3 gblk(D_MODELc / 64, MTOT / 128);   // (16, 32)
    dim3 blk(256);

    proj_kernel<0><<<gblk, blk, 0, stream>>>(q, Wqb, bq, Qp);
    proj_kernel<0><<<gblk, blk, 0, stream>>>(k, Wkb, bk, Kp);
    proj_kernel<1><<<gblk, blk, 0, stream>>>(v, Wvb, bv, Vt);

    attn_kernel<<<dim3(BATCHc * HEADSc * (SEQc / 64)), dim3(128), 0, stream>>>(Qp, Kp, Vt, Ao);

    outproj_kernel<<<gblk, blk, 0, stream>>>(Ao, Wob, bo, out);
}
